// FixedGraphTransformer2DModel_30227979829642
// MI455X (gfx1250) — matrix operand order repaired and checked by reading, not compiled, not run on hardware
//
#include <hip/hip_runtime.h>
#include <hip/hip_bf16.h>

// ---------------------------------------------------------------------------
// Problem constants (match reference)
// ---------------------------------------------------------------------------
#define NTOT_C   8320          // B*NODES nodes
#define E_C      266240        // edges
#define S_C      64            // seq length
#define D_C      128           // model dim
#define H_C      8             // heads
#define HD_C     16            // head dim
#define DFF_C    256
#define BN_S_C   (NTOT_C * S_C)   // 532480 token rows
#define NODES_C  130

typedef __bf16 bf16_t;
typedef __attribute__((ext_vector_type(16))) __bf16 v16bf;
typedef __attribute__((ext_vector_type(8)))  float  v8f;

// ---------------------------------------------------------------------------
// CDNA5 async global->LDS path (ASYNCcnt-tracked DMA writes to LDS)
// clang-22 prototype: (global v4i* src, local v4i* dst, int offset, int cpol)
// ---------------------------------------------------------------------------
#if defined(__has_builtin)
#  if __has_builtin(__builtin_amdgcn_global_load_async_to_lds_b128)
#    define HAVE_ASYNC_LDS 1
#  endif
#endif

static __device__ __forceinline__ void wait_async0() {
#if defined(__has_builtin) && __has_builtin(__builtin_amdgcn_s_wait_asynccnt)
    __builtin_amdgcn_s_wait_asynccnt(0);
#else
    asm volatile("s_wait_asynccnt 0" ::: "memory");
#endif
}

#ifdef HAVE_ASYNC_LDS
typedef int v4i_gcc __attribute__((vector_size(16)));
typedef __attribute__((address_space(1))) v4i_gcc* gp_v4i;
typedef __attribute__((address_space(3))) v4i_gcc* lp_v4i;
static __device__ __forceinline__ void async_cp16(const void* g, void* l) {
    __builtin_amdgcn_global_load_async_to_lds_b128((gp_v4i)(void*)g, (lp_v4i)l, 0, 0);
}
#endif

// ---------------------------------------------------------------------------
// bf16 <-> f32 helpers (bit-exact RNE, no libcalls)
// ---------------------------------------------------------------------------
static __device__ __forceinline__ bf16_t f2b(float f) {
    unsigned u = __builtin_bit_cast(unsigned, f);
    unsigned r = (u + 0x7FFFu + ((u >> 16) & 1u)) >> 16;
    return __builtin_bit_cast(__bf16, (unsigned short)r);
}
static __device__ __forceinline__ float b2f(bf16_t h) {
    unsigned short s = __builtin_bit_cast(unsigned short, h);
    unsigned u = ((unsigned)s) << 16;
    return __builtin_bit_cast(float, u);
}

// ---------------------------------------------------------------------------
// WMMA helpers (CDNA5: v_wmma_f32_16x16x32_bf16, wave32)
// A (16x32, bf16): lanes 0-15 rows M=0..15 (K 0..7,16..23), lanes 16-31 same
//                  rows with K 8..15,24..31.
// B (32x16, bf16): lane&15 = N col, K split same as A (lanes 0-15 K 0..7,16..23).
// C/D (16x16 f32): VGPR r -> M = r + 8*(lane>=16), N = lane&15.
// ---------------------------------------------------------------------------
static __device__ __forceinline__ v8f wmma_bf16(v16bf a, v16bf b, v8f c) {
    return __builtin_amdgcn_wmma_f32_16x16x32_bf16(
        /*neg_a=*/false, a, /*neg_b=*/false, b,
        /*c_mod=*/(short)0, c, /*reuse_a=*/false, /*reuse_b=*/false);
}

static __device__ __forceinline__ v16bf load_a_frag(const bf16_t* A, int lda, int lane) {
    const int row = lane & 15;
    const int kb  = (lane >> 4) << 3;
    const bf16_t* p = A + row * lda + kb;
    v16bf a;
#pragma unroll
    for (int i = 0; i < 8; ++i) { a[i] = p[i]; a[i + 8] = p[i + 16]; }
    return a;
}

// A fragment for K=16 padded to 32 (upper half zero)
static __device__ __forceinline__ v16bf load_a_frag_k16(const bf16_t* A, int lda, int lane) {
    const int row = lane & 15;
    const int kb  = (lane >> 4) << 3;
    const bf16_t* p = A + row * lda + kb;
    const bf16_t z = __builtin_bit_cast(__bf16, (unsigned short)0);
    v16bf a;
#pragma unroll
    for (int i = 0; i < 8; ++i) { a[i] = p[i]; a[i + 8] = z; }
    return a;
}

static __device__ __forceinline__ v16bf load_b_frag(const bf16_t* B, int ldb, int lane) {
    const bf16_t* p = B + ((lane >> 4) << 3) * ldb + (lane & 15);
    v16bf b;
#pragma unroll
    for (int i = 0; i < 8; ++i) { b[i] = p[i * ldb]; b[i + 8] = p[(i + 16) * ldb]; }
    return b;
}

// ---------------------------------------------------------------------------
// Generic WMMA GEMM:  out(bf16)[M,N] = A(bf16)[M,K] @ Bw(bf16)[K,N] + bias
//                     (+ residual bf16)  (+ ReLU)
// block = 256 thr (8 waves), tile 128(M) x 128(N), K step 32.
// B tile staged in LDS via async global->LDS DMA when available.
// ---------------------------------------------------------------------------
template <bool RELU, bool HAS_RES>
__global__ __launch_bounds__(256) void gemm_bf16_kernel(
    const bf16_t* __restrict__ A, const bf16_t* __restrict__ Bw,
    const float* __restrict__ bias, const bf16_t* __restrict__ res,
    bf16_t* __restrict__ out, int M, int N, int K)
{
    __shared__ bf16_t Bs[32 * 128];   // 8KB
    const int tid  = threadIdx.x;
    const int lane = tid & 31;
    const int wid  = tid >> 5;
    const int m0   = blockIdx.x * 128 + wid * 16;
    const int n0   = blockIdx.y * 128;

    v8f acc[8] = {};

    for (int kk = 0; kk < K; kk += 32) {
        // cooperative stage of B tile [32][128]: 512 x 16B chunks, 2 per thread
#pragma unroll
        for (int c = 0; c < 2; ++c) {
            const int ci = tid * 2 + c;
            const int kr = ci >> 4;
            const int nc = (ci & 15) << 3;
            const bf16_t* src = Bw + (size_t)(kk + kr) * N + n0 + nc;
            bf16_t* dst = Bs + kr * 128 + nc;
#ifdef HAVE_ASYNC_LDS
            async_cp16(src, dst);
#else
            *(uint4*)dst = *(const uint4*)src;
#endif
        }
#ifdef HAVE_ASYNC_LDS
        wait_async0();
#endif
        __syncthreads();

        if (kk + 32 < K)
            __builtin_prefetch(A + (size_t)m0 * K + kk + 32, 0, 0); // global_prefetch_b8

        v16bf a = load_a_frag(A + (size_t)m0 * K + kk, K, lane);
#pragma unroll
        for (int t = 0; t < 8; ++t) {
            v16bf b = load_b_frag(Bs + t * 16, 128, lane);
            acc[t] = wmma_bf16(a, b, acc[t]);
        }
        __syncthreads();
    }

#pragma unroll
    for (int t = 0; t < 8; ++t) {
#pragma unroll
        for (int r = 0; r < 8; ++r) {
            const int mm = m0 + r + ((lane >> 4) << 3);
            const int nn = n0 + t * 16 + (lane & 15);
            float v = acc[t][r] + bias[nn];
            if (HAS_RES) v += b2f(res[(size_t)mm * N + nn]);
            if (RELU)    v = v > 0.0f ? v : 0.0f;
            out[(size_t)mm * N + nn] = f2b(v);
        }
    }
}

// ---------------------------------------------------------------------------
// Fused temporal attention: one block = (one sequence of 64 tokens, 2 heads)
// grid = (BN, 4), block = 128 (4 waves). Everything via WMMA.
// ---------------------------------------------------------------------------
__global__ __launch_bounds__(128) void temporal_attn_kernel(
    const bf16_t* __restrict__ ht,    // [BN*S, D] bf16 (h = x + pe)
    const bf16_t* __restrict__ Winb,  // [D, 3D]  bf16
    const float*  __restrict__ b_in,  // [3D]
    bf16_t* __restrict__ attnout)     // [BN*S, D] bf16
{
    __shared__ bf16_t hs[64 * 128];   // 16KB token tile
    __shared__ bf16_t qs[64 * 32];    // 4KB  (2 heads)
    __shared__ bf16_t ks_[64 * 32];   // 4KB
    __shared__ bf16_t vs[64 * 32];    // 4KB
    __shared__ float  sc[64 * 64];    // 16KB scores (one head at a time)
    __shared__ bf16_t at[64 * 64];    // 8KB  attn weights bf16

    const int seq  = blockIdx.x;
    const int hp   = blockIdx.y;      // head pair 0..3
    const int tid  = threadIdx.x;
    const int lane = tid & 31;
    const int wid  = tid >> 5;
    const int m0   = wid * 16;

    // load token tile (16KB, async DMA to LDS when available)
    {
        const uint4* src = (const uint4*)(ht + (size_t)seq * 64 * 128);
        uint4* dst = (uint4*)hs;
#pragma unroll
        for (int i = 0; i < 8; ++i) {
#ifdef HAVE_ASYNC_LDS
            async_cp16(src + tid + i * 128, dst + tid + i * 128);
#else
            dst[tid + i * 128] = src[tid + i * 128];
#endif
        }
    }
#ifdef HAVE_ASYNC_LDS
    wait_async0();
#endif
    __syncthreads();

    // ---- Q/K/V for this head pair: [64,128] @ [128,32] ----
#pragma unroll
    for (int m = 0; m < 3; ++m) {
        const int colofs = m * 128 + hp * 32;
        v8f acc[2] = {};
        for (int kk = 0; kk < 128; kk += 32) {
            v16bf a = load_a_frag(hs + m0 * 128 + kk, 128, lane);
#pragma unroll
            for (int t = 0; t < 2; ++t) {
                v16bf b;
                const bf16_t* p = Winb + (size_t)(kk + ((lane >> 4) << 3)) * 384 + colofs + t * 16 + (lane & 15);
#pragma unroll
                for (int i = 0; i < 8; ++i) { b[i] = p[i * 384]; b[i + 8] = p[(i + 16) * 384]; }
                acc[t] = wmma_bf16(a, b, acc[t]);
            }
        }
        bf16_t* dst = (m == 0) ? qs : ((m == 1) ? ks_ : vs);
#pragma unroll
        for (int t = 0; t < 2; ++t)
#pragma unroll
            for (int r = 0; r < 8; ++r) {
                const int mm = r + ((lane >> 4) << 3);
                const int nn = t * 16 + (lane & 15);
                dst[(m0 + mm) * 32 + nn] = f2b(acc[t][r] + b_in[colofs + nn]);
            }
    }
    __syncthreads();

    const bf16_t zb = __builtin_bit_cast(__bf16, (unsigned short)0);

    for (int it = 0; it < 2; ++it) {
        const int hc = it * 16;

        // ---- scores = q[64,16(pad32)] @ k^T[16,64], scaled ----
        {
            v16bf a = load_a_frag_k16(qs + m0 * 32 + hc, 32, lane);
#pragma unroll
            for (int t = 0; t < 4; ++t) {
                v16bf b;
#pragma unroll
                for (int i = 0; i < 8; ++i) { b[i] = ks_[(t * 16 + (lane & 15)) * 32 + hc + ((lane >> 4) << 3) + i]; b[i + 8] = zb; }
                v8f z = {};
                v8f s = wmma_bf16(a, b, z);
#pragma unroll
                for (int r = 0; r < 8; ++r) {
                    const int mm = r + ((lane >> 4) << 3);
                    const int nn = t * 16 + (lane & 15);
                    sc[(m0 + mm) * 64 + nn] = s[r] * 0.25f; // 1/sqrt(16)
                }
            }
        }
        __syncthreads();

        // ---- softmax rows (64 threads, one row each) ----
        if (tid < 64) {
            float mx = -1e30f;
            for (int j = 0; j < 64; ++j) mx = fmaxf(mx, sc[tid * 64 + j]);
            float sum = 0.0f;
            for (int j = 0; j < 64; ++j) {
                float e = __expf(sc[tid * 64 + j] - mx);
                sc[tid * 64 + j] = e;
                sum += e;
            }
            const float rs = 1.0f / (sum + 1e-16f);
            for (int j = 0; j < 64; ++j) at[tid * 64 + j] = f2b(sc[tid * 64 + j] * rs);
        }
        __syncthreads();

        // ---- o = attn[64,64] @ v[64,16] ----
        {
            v8f acc = {};
#pragma unroll
            for (int s2 = 0; s2 < 2; ++s2) {
                v16bf a = load_a_frag(at + m0 * 64 + s2 * 32, 64, lane);
                v16bf b;
#pragma unroll
                for (int i = 0; i < 8; ++i) { b[i] = vs[(s2 * 32 + ((lane >> 4) << 3) + i) * 32 + hc + (lane & 15)]; b[i + 8] = vs[(s2 * 32 + 16 + ((lane >> 4) << 3) + i) * 32 + hc + (lane & 15)]; }
                acc = wmma_bf16(a, b, acc);
            }
#pragma unroll
            for (int r = 0; r < 8; ++r) {
                const int mm = r + ((lane >> 4) << 3);
                const int nn = lane & 15;
                attnout[((size_t)seq * 64 + m0 + mm) * 128 + hp * 32 + hc + nn] = f2b(acc[r]);
            }
        }
        __syncthreads();
    }
}

// ---------------------------------------------------------------------------
// LayerNorm over 128 cols (one wave per row, 8 rows per block)
// ---------------------------------------------------------------------------
__global__ __launch_bounds__(256) void ln_kernel(
    const bf16_t* __restrict__ in, const float* __restrict__ g,
    const float* __restrict__ bta, bf16_t* __restrict__ out, int M)
{
    const int row  = blockIdx.x * 8 + (threadIdx.x >> 5);
    const int lane = threadIdx.x & 31;
    if (row >= M) return;
    const bf16_t* p = in + (size_t)row * 128;
    float x[4];
#pragma unroll
    for (int i = 0; i < 4; ++i) x[i] = b2f(p[lane + 32 * i]);
    float s = x[0] + x[1] + x[2] + x[3];
    for (int o = 16; o > 0; o >>= 1) s += __shfl_xor(s, o, 32);
    const float mu = s * (1.0f / 128.0f);
    float v2 = 0.0f;
#pragma unroll
    for (int i = 0; i < 4; ++i) { const float d = x[i] - mu; v2 += d * d; }
    for (int o = 16; o > 0; o >>= 1) v2 += __shfl_xor(v2, o, 32);
    const float inv = rsqrtf(v2 * (1.0f / 128.0f) + 1e-5f);
#pragma unroll
    for (int i = 0; i < 4; ++i) {
        const int c = lane + 32 * i;
        out[(size_t)row * 128 + c] = f2b((x[i] - mu) * inv * g[c] + bta[c]);
    }
}

// ---------------------------------------------------------------------------
// Small elementwise / packing kernels
// ---------------------------------------------------------------------------
__global__ void cvt_kernel(const float* __restrict__ in, bf16_t* __restrict__ out, int n) {
    const int i = blockIdx.x * 256 + threadIdx.x;
    if (i < n) out[i] = f2b(in[i]);
}

__global__ void zero_kernel(float* __restrict__ p, int n) {
    const int i = blockIdx.x * 256 + threadIdx.x;
    if (i < n) p[i] = 0.0f;
}

// out = in + spatial_pe[node % 130]
__global__ void spe_add_kernel(const bf16_t* __restrict__ in, const float* __restrict__ spe,
                               bf16_t* __restrict__ out) {
    const int idx = blockIdx.x * 256 + threadIdx.x; // NTOT*128
    if (idx >= NTOT_C * 128) return;
    const int d = idx & 127;
    const int n = idx >> 7;
    const int node = n % NODES_C;
    out[idx] = f2b(b2f(in[idx]) + spe[node * 128 + d]);
}

// ht[bn,s,d] = in + temporal_pe[s];  bcast: in indexed [bn,d] (graph output)
__global__ void pe_add_kernel(const bf16_t* __restrict__ in, const float* __restrict__ pe,
                              bf16_t* __restrict__ out, int bcast) {
    const size_t idx = (size_t)blockIdx.x * 256 + threadIdx.x; // BN_S*128
    const int d = (int)(idx & 127);
    const int s = (int)((idx >> 7) & 63);
    const size_t bn = idx >> 13;
    const float v = b2f(bcast ? in[bn * 128 + d] : in[idx]);
    out[idx] = f2b(v + pe[s * 128 + d]);
}

// Pack graph per-head weights [H,D,HD] x4 -> Wcat bf16 [D, 4D] and bcat f32 [4D]
__global__ void pack_graph_kernel(const float* __restrict__ Wq, const float* __restrict__ Wk,
                                  const float* __restrict__ Wv, const float* __restrict__ Ws,
                                  const float* __restrict__ bq, const float* __restrict__ bk,
                                  const float* __restrict__ bv, const float* __restrict__ bs,
                                  bf16_t* __restrict__ Wcat, float* __restrict__ bcat) {
    const int idx = blockIdx.x * 256 + threadIdx.x;
    if (idx < 128 * 512) {
        const int d = idx >> 9;
        const int c = idx & 511;
        const int m = c >> 7, hh = (c >> 4) & 7, e = c & 15;
        const float* W = (m == 0) ? Wq : ((m == 1) ? Wk : ((m == 2) ? Wv : Ws));
        Wcat[d * 512 + c] = f2b(W[hh * (128 * 16) + d * 16 + e]);
    }
    if (idx < 512) {
        const int m = idx >> 7, c = idx & 127;
        const float* bb = (m == 0) ? bq : ((m == 1) ? bk : ((m == 2) ? bv : bs));
        bcat[idx] = bb[c];
    }
}

// Edge attention scatter: exp(logit) accumulated per destination node
__global__ void edge_scatter_kernel(const int* __restrict__ ei, const float* __restrict__ ea,
                                    const float* __restrict__ We, const bf16_t* __restrict__ qkvs,
                                    float* __restrict__ num, float* __restrict__ den) {
    const long long idx = (long long)blockIdx.x * 256 + threadIdx.x;
    if (idx >= (long long)E_C * 8) return;
    const int e = (int)(idx >> 3);
    const int h = (int)(idx & 7);
    const int srcn = ei[e];
    const int dstn = ei[E_C + e];
    const float eav = ea[e];
    const bf16_t* qd = qkvs + (size_t)dstn * 512 + h * 16;
    const bf16_t* kp = qkvs + (size_t)srcn * 512 + 128 + h * 16;
    const bf16_t* vp = qkvs + (size_t)srcn * 512 + 256 + h * 16;
    const float* we = We + h * 16;
    float logit = 0.0f;
#pragma unroll
    for (int i = 0; i < 16; ++i) logit += b2f(qd[i]) * (b2f(kp[i]) + eav * we[i]);
    const float ex = __expf(logit * 0.25f); // 1/sqrt(HD)
    atomicAdd(den + (size_t)dstn * 8 + h, ex);
#pragma unroll
    for (int i = 0; i < 16; ++i)
        atomicAdd(num + (size_t)dstn * 128 + h * 16 + i, ex * (b2f(vp[i]) + eav * we[i]));
}

// normalize + add skip branch -> aggb bf16
__global__ void graph_norm_kernel(const float* __restrict__ num, const float* __restrict__ den,
                                  const bf16_t* __restrict__ qkvs, bf16_t* __restrict__ aggb) {
    const int idx = blockIdx.x * 256 + threadIdx.x; // NTOT*128
    if (idx >= NTOT_C * 128) return;
    const int n = idx >> 7;
    const int c = idx & 127;
    const int h = c >> 4;
    const float v = num[idx] / (den[n * 8 + h] + 1e-16f) + b2f(qkvs[(size_t)n * 512 + 384 + c]);
    aggb[idx] = f2b(v);
}

// mean over NODES*S per batch -> pooled f32 [B, D]
__global__ void pool_kernel(const bf16_t* __restrict__ xin, float* __restrict__ pooled) {
    const int b = blockIdx.x;   // 64
    const int d = threadIdx.x;  // 128
    float s = 0.0f;
    const bf16_t* p = xin + (size_t)b * 8320 * 128 + d;
    for (int r = 0; r < 8320; ++r) s += b2f(p[(size_t)r * 128]);
    pooled[b * 128 + d] = s * (1.0f / 8320.0f);
}

// tiny MLP head: 128 -> 64 -> 32 -> 1
__global__ void head_kernel(const float* __restrict__ pooled,
                            const float* __restrict__ W1, const float* __restrict__ b1,
                            const float* __restrict__ W2, const float* __restrict__ b2,
                            const float* __restrict__ W3, const float* __restrict__ b3,
                            float* __restrict__ out) {
    const int b = threadIdx.x;
    if (b >= 64) return;
    float z1[64];
    for (int j = 0; j < 64; ++j) {
        float s = b1[j];
        for (int k2 = 0; k2 < 128; ++k2) s += pooled[b * 128 + k2] * W1[k2 * 64 + j];
        z1[j] = s > 0.0f ? s : 0.0f;
    }
    float z2[32];
    for (int j = 0; j < 32; ++j) {
        float s = b2[j];
        for (int k2 = 0; k2 < 64; ++k2) s += z1[k2] * W2[k2 * 32 + j];
        z2[j] = s > 0.0f ? s : 0.0f;
    }
    float s = b3[0];
    for (int k2 = 0; k2 < 32; ++k2) s += z2[k2] * W3[k2];
    out[b] = s;
}

// ---------------------------------------------------------------------------
// Host launch sequence
// ---------------------------------------------------------------------------
static inline size_t alup(size_t x) { return (x + 255) & ~(size_t)255; }

extern "C" void kernel_launch(void* const* d_in, const int* in_sizes, int n_in,
                              void* d_out, int out_size, void* d_ws, size_t ws_size,
                              hipStream_t stream) {
    (void)in_sizes; (void)n_in; (void)out_size; (void)ws_size;

    // ---- input pointers (setup_inputs order, params flattened depth-first) ----
    const float* x_in = (const float*)d_in[0];
    const int*   ei   = (const int*)d_in[1];
    const float* ea   = (const float*)d_in[2];
    // d_in[3] = batch_size (constant 64)
    const float* in_W = (const float*)d_in[4];
    const float* in_b = (const float*)d_in[5];
    const float* spe  = (const float*)d_in[6];
    const float* tpe  = (const float*)d_in[7];
    const int GP0 = 8;                 // graph layer params: 13 each
    const int TP0 = 8 + 2 * 13;       // temporal layer params: 12 each
    const int HP0 = TP0 + 2 * 12;     // head params

    // ---- workspace arena ----
    char* ws = (char*)d_ws;
    size_t off = 0;
    auto take = [&](size_t bytes) { char* p = ws + off; off += alup(bytes); return p; };

    bf16_t* xb    = (bf16_t*)take((size_t)NTOT_C * 64 * 2);
    bf16_t* inWb  = (bf16_t*)take(64 * 128 * 2);
    bf16_t* hA    = (bf16_t*)take((size_t)NTOT_C * 128 * 2);
    bf16_t* hB    = (bf16_t*)take((size_t)NTOT_C * 128 * 2);
    bf16_t* preb  = (bf16_t*)take((size_t)NTOT_C * 128 * 2);
    bf16_t* qkvs  = (bf16_t*)take((size_t)NTOT_C * 512 * 2);
    bf16_t* aggb  = (bf16_t*)take((size_t)NTOT_C * 128 * 2);
    float*  num   = (float*)take((size_t)NTOT_C * 128 * 4);
    float*  den   = (float*)take((size_t)NTOT_C * 8 * 4);
    bf16_t* Wcatb = (bf16_t*)take(128 * 512 * 2);
    float*  bcat  = (float*)take(512 * 4);
    bf16_t* Wogb  = (bf16_t*)take(128 * 128 * 2);
    bf16_t* Winb  = (bf16_t*)take(128 * 384 * 2);
    bf16_t* Wotb  = (bf16_t*)take(128 * 128 * 2);
    bf16_t* W1b   = (bf16_t*)take(128 * 256 * 2);
    bf16_t* W2b   = (bf16_t*)take(256 * 128 * 2);
    float*  pooled= (float*)take(64 * 128 * 4);
    bf16_t* T0    = (bf16_t*)take((size_t)BN_S_C * 128 * 2);
    bf16_t* T1    = (bf16_t*)take((size_t)BN_S_C * 128 * 2);
    bf16_t* T2    = (bf16_t*)take((size_t)BN_S_C * 128 * 2);
    bf16_t* T3    = (bf16_t*)take((size_t)BN_S_C * 256 * 2);

    const dim3 blk256(256), blk128(128);
    auto cvt = [&](const float* src, bf16_t* dst, int n) {
        cvt_kernel<<<dim3((n + 255) / 256), blk256, 0, stream>>>(src, dst, n);
    };

    // ---- Stage A: input projection + spatial PE ----
    cvt(x_in, xb, NTOT_C * 64);
    cvt(in_W, inWb, 64 * 128);
    gemm_bf16_kernel<false, false><<<dim3(NTOT_C / 128, 1), blk256, 0, stream>>>(
        xb, inWb, in_b, nullptr, preb, NTOT_C, 128, 64);
    spe_add_kernel<<<dim3(NTOT_C * 128 / 256), blk256, 0, stream>>>(preb, spe, hA);

    // ---- Graph layers ----
    bf16_t* hcur = hA;
    bf16_t* hnext = hB;
    for (int l = 0; l < 2; ++l) {
        const int gi = GP0 + l * 13;
        const float* Wq = (const float*)d_in[gi + 0];
        const float* bq = (const float*)d_in[gi + 1];
        const float* Wk = (const float*)d_in[gi + 2];
        const float* bk = (const float*)d_in[gi + 3];
        const float* Wv = (const float*)d_in[gi + 4];
        const float* bv = (const float*)d_in[gi + 5];
        const float* We = (const float*)d_in[gi + 6];
        const float* Wsk= (const float*)d_in[gi + 7];
        const float* bsk= (const float*)d_in[gi + 8];
        const float* Wo = (const float*)d_in[gi + 9];
        const float* bo = (const float*)d_in[gi + 10];
        const float* lng= (const float*)d_in[gi + 11];
        const float* lnb= (const float*)d_in[gi + 12];

        pack_graph_kernel<<<dim3(128 * 512 / 256), blk256, 0, stream>>>(
            Wq, Wk, Wv, Wsk, bq, bk, bv, bsk, Wcatb, bcat);
        cvt(Wo, Wogb, 128 * 128);

        gemm_bf16_kernel<false, false><<<dim3(NTOT_C / 128, 512 / 128), blk256, 0, stream>>>(
            hcur, Wcatb, bcat, nullptr, qkvs, NTOT_C, 512, 128);

        zero_kernel<<<dim3(NTOT_C * 128 / 256), blk256, 0, stream>>>(num, NTOT_C * 128);
        zero_kernel<<<dim3((NTOT_C * 8 + 255) / 256), blk256, 0, stream>>>(den, NTOT_C * 8);
        edge_scatter_kernel<<<dim3((int)(((long long)E_C * 8) / 256)), blk256, 0, stream>>>(
            ei, ea, We, qkvs, num, den);
        graph_norm_kernel<<<dim3(NTOT_C * 128 / 256), blk256, 0, stream>>>(num, den, qkvs, aggb);

        gemm_bf16_kernel<false, true><<<dim3(NTOT_C / 128, 1), blk256, 0, stream>>>(
            aggb, Wogb, bo, hcur, preb, NTOT_C, 128, 128);
        ln_kernel<<<dim3(NTOT_C / 8), blk256, 0, stream>>>(preb, lng, lnb, hnext, NTOT_C);

        bf16_t* t = hcur; hcur = hnext; hnext = t;
    }

    // ---- Temporal layers ----
    const int peGrid = (int)((size_t)BN_S_C * 128 / 256);
    for (int l = 0; l < 2; ++l) {
        const int ti = TP0 + l * 12;
        const float* W_in = (const float*)d_in[ti + 0];
        const float* b_in = (const float*)d_in[ti + 1];
        const float* Wo   = (const float*)d_in[ti + 2];
        const float* bo   = (const float*)d_in[ti + 3];
        const float* W1   = (const float*)d_in[ti + 4];
        const float* b1   = (const float*)d_in[ti + 5];
        const float* W2   = (const float*)d_in[ti + 6];
        const float* b2   = (const float*)d_in[ti + 7];
        const float* ln1g = (const float*)d_in[ti + 8];
        const float* ln1b = (const float*)d_in[ti + 9];
        const float* ln2g = (const float*)d_in[ti + 10];
        const float* ln2b = (const float*)d_in[ti + 11];

        cvt(W_in, Winb, 128 * 384);
        cvt(Wo, Wotb, 128 * 128);
        cvt(W1, W1b, 128 * 256);
        cvt(W2, W2b, 256 * 128);

        // h = x + pe  (layer 0 broadcasts the graph output over S)
        const bf16_t* xin = (l == 0) ? hcur : T2;
        pe_add_kernel<<<dim3(peGrid), blk256, 0, stream>>>(xin, tpe, T0, (l == 0) ? 1 : 0);

        // self-attention -> T1
        temporal_attn_kernel<<<dim3(NTOT_C, 4), blk128, 0, stream>>>(T0, Winb, b_in, T1);

        // o @ Wo + bo + h  -> T3 (pre-LN1), LN1 -> T1
        gemm_bf16_kernel<false, true><<<dim3(BN_S_C / 128, 1), blk256, 0, stream>>>(
            T1, Wotb, bo, T0, T3, BN_S_C, 128, 128);
        ln_kernel<<<dim3(BN_S_C / 8), blk256, 0, stream>>>(T3, ln1g, ln1b, T1, BN_S_C);

        // FFN: relu(T1 @ W1 + b1) -> T3 ; T3 @ W2 + b2 + T1 -> T0 ; LN2 -> T2
        gemm_bf16_kernel<true, false><<<dim3(BN_S_C / 128, 256 / 128), blk256, 0, stream>>>(
            T1, W1b, b1, nullptr, T3, BN_S_C, 256, 128);
        gemm_bf16_kernel<false, true><<<dim3(BN_S_C / 128, 1), blk256, 0, stream>>>(
            T3, W2b, b2, T1, T0, BN_S_C, 128, 256);
        ln_kernel<<<dim3(BN_S_C / 8), blk256, 0, stream>>>(T0, ln2g, ln2b, T2, BN_S_C);
    }

    // ---- Pool + head ----
    pool_kernel<<<dim3(64), blk128, 0, stream>>>(T2, pooled);
    head_kernel<<<dim3(1), dim3(64), 0, stream>>>(
        pooled,
        (const float*)d_in[HP0 + 0], (const float*)d_in[HP0 + 1],
        (const float*)d_in[HP0 + 2], (const float*)d_in[HP0 + 3],
        (const float*)d_in[HP0 + 4], (const float*)d_in[HP0 + 5],
        (float*)d_out);
}
